// ManagedVanillaRNN_24481313587730
// MI455X (gfx1250) — compile-verified, matrix operand order checked by reading, not run on hardware
//
#include <hip/hip_runtime.h>

// Problem sizes (match reference).
static constexpr int Bn   = 32;
static constexpr int Tn   = 512;
static constexpr int INn  = 512;
static constexpr int Hn   = 1024;
static constexpr int OUTn = 512;

typedef __attribute__((ext_vector_type(16))) __bf16 v16bf;
typedef __attribute__((ext_vector_type(8)))  float  v8f;

union FragBF {
    v16bf v;
    uint4 q[2];
    unsigned short s[16];
};

__device__ __forceinline__ unsigned short f2bf(float f) {
    unsigned int u = __float_as_uint(f);
    u += 0x7FFFu + ((u >> 16) & 1u);   // round-to-nearest-even
    return (unsigned short)(u >> 16);
}

// Load a 16x32 bf16 A/B fragment from a bf16 row pointer.
// Lane layout (ISA 7.12.2): lanes 0-15 hold K = {kBase..+7, kBase+16..+23},
// lanes 16-31 hold K = {kBase+8..+15, kBase+24..+31}.
__device__ __forceinline__ v16bf load_frag_bf16(const unsigned short* rowPtr,
                                                int kBase, int half) {
    FragBF f;
    f.q[0] = *reinterpret_cast<const uint4*>(rowPtr + kBase + 8 * half);
    f.q[1] = *reinterpret_cast<const uint4*>(rowPtr + kBase + 16 + 8 * half);
    return f.v;
}

__device__ __forceinline__ v8f wmma_bf16(v16bf a, v16bf b, v8f c) {
    return __builtin_amdgcn_wmma_f32_16x16x32_bf16(
        /*neg_a=*/false, a, /*neg_b=*/false, b,
        /*c_mod=*/(short)0, c, /*reuse_a=*/false, /*reuse_b=*/false);
}

// ---------------------------------------------------------------------------
// Prep kernels: weight transpose+convert to bf16; straight convert for x / h0.
// dst[n*K + k] = bf16(src[k*N + n])
__global__ __launch_bounds__(256) void transpose_cvt_kernel(const float* __restrict__ src,
                                                            unsigned short* __restrict__ dst,
                                                            int K, int N) {
    int idx = blockIdx.x * 256 + threadIdx.x;
    if (idx < K * N) {
        int k = idx / N, n = idx % N;
        dst[(size_t)n * K + k] = f2bf(src[idx]);
    }
}

__global__ __launch_bounds__(256) void cvt_kernel(const float* __restrict__ src,
                                                  unsigned short* __restrict__ dst,
                                                  int n) {
    int idx = blockIdx.x * 256 + threadIdx.x;
    if (idx < n) dst[idx] = f2bf(src[idx]);
}

// ---------------------------------------------------------------------------
// GEMM1: P[(t*B + b)*H + n] = x[b*T+t, :] @ Wx[:, n] + bx[n] + bh[n]
// A = x_bf16 [B*T][IN], B = WxT bf16 [H][IN]. 64x32 tile per wave:
// steady state 8 WMMA : 6 fragment loads.
__global__ __launch_bounds__(256) void gemm_xwx_kernel(const unsigned short* __restrict__ xb,
                                                       const unsigned short* __restrict__ WxT,
                                                       const float* __restrict__ bx,
                                                       const float* __restrict__ bh,
                                                       float* __restrict__ P) {
    const int lane = threadIdx.x & 31;
    const int wave = threadIdx.x >> 5;
    const int half = lane >> 4;
    const int lane16 = lane & 15;

    const int nTN = Hn / 32;                       // 32 N-tiles
    const int wid = blockIdx.x * 8 + wave;         // 8192 wave tiles
    const int tm = wid / nTN, tn = wid % nTN;
    const int m0 = tm * 64, n0 = tn * 32;

    const unsigned short* arow[4];
    for (int mi = 0; mi < 4; ++mi)
        arow[mi] = xb + (size_t)(m0 + mi * 16 + lane16) * INn;
    const unsigned short* brow0 = WxT + (size_t)(n0 + lane16) * INn;
    const unsigned short* brow1 = WxT + (size_t)(n0 + 16 + lane16) * INn;

    v8f acc[4][2];
    for (int i = 0; i < 4; ++i)
        for (int j = 0; j < 2; ++j)
            for (int r = 0; r < 8; ++r) acc[i][j][r] = 0.0f;

    for (int kc = 0; kc < INn; kc += 32) {
        v16bf b0 = load_frag_bf16(brow0, kc, half);
        v16bf b1 = load_frag_bf16(brow1, kc, half);
        for (int mi = 0; mi < 4; ++mi) {
            v16bf a = load_frag_bf16(arow[mi], kc, half);
            acc[mi][0] = wmma_bf16(a, b0, acc[mi][0]);
            acc[mi][1] = wmma_bf16(a, b1, acc[mi][1]);
        }
    }

    for (int mi = 0; mi < 4; ++mi) {
        for (int ni = 0; ni < 2; ++ni) {
            int n = n0 + ni * 16 + lane16;
            float bias = bx[n] + bh[n];
            for (int r = 0; r < 8; ++r) {
                int g = m0 + mi * 16 + r + 8 * half;   // row in [0, B*T): g = b*T + t
                int bb = g >> 9;                       // / T
                int tt = g & (Tn - 1);                 // % T
                P[((size_t)tt * Bn + bb) * Hn + n] = acc[mi][ni][r] + bias;
            }
        }
    }
}

// ---------------------------------------------------------------------------
// One RNN step: h_new = tanh(P_t + h_prev @ Wh). 64 waves, each owns 32x16
// (latency-bound; maximize wave-level parallelism, not per-wave intensity).
__global__ __launch_bounds__(128) void rnn_step_kernel(const unsigned short* __restrict__ hprev,
                                                       long long hstride,
                                                       const unsigned short* __restrict__ WhT,
                                                       const float* __restrict__ Pt,
                                                       unsigned short* __restrict__ outs,
                                                       int t,
                                                       float* __restrict__ nh) {
    const int lane = threadIdx.x & 31;
    const int wave = threadIdx.x >> 5;
    const int half = lane >> 4;
    const int lane16 = lane & 15;

    const int wid = blockIdx.x * 4 + wave;   // 0..63
    const int n0 = wid * 16;

    const unsigned short* arow0 = hprev + (size_t)lane16 * hstride;
    const unsigned short* arow1 = hprev + (size_t)(lane16 + 16) * hstride;
    const unsigned short* brow  = WhT + (size_t)(n0 + lane16) * Hn;

    v8f acc0, acc1;
    for (int r = 0; r < 8; ++r) { acc0[r] = 0.0f; acc1[r] = 0.0f; }

    for (int kc = 0; kc < Hn; kc += 32) {
        v16bf a0 = load_frag_bf16(arow0, kc, half);
        v16bf a1 = load_frag_bf16(arow1, kc, half);
        v16bf b  = load_frag_bf16(brow, kc, half);
        acc0 = wmma_bf16(a0, b, acc0);
        acc1 = wmma_bf16(a1, b, acc1);
    }

    const int n = n0 + lane16;
    for (int r = 0; r < 8; ++r) {
        int b0 = r + 8 * half;            // batch rows 0..15
        int b1 = 16 + r + 8 * half;       // batch rows 16..31
        float v0 = tanhf(acc0[r] + Pt[(size_t)b0 * Hn + n]);
        float v1 = tanhf(acc1[r] + Pt[(size_t)b1 * Hn + n]);
        outs[((size_t)b0 * Tn + t) * Hn + n] = f2bf(v0);
        outs[((size_t)b1 * Tn + t) * Hn + n] = f2bf(v1);
        if (nh) {
            nh[(size_t)b0 * Hn + n] = v0;
            nh[(size_t)b1 * Hn + n] = v1;
        }
    }
}

// ---------------------------------------------------------------------------
// td GEMMs: dout[z][(b*T+t)*OUT + n] = outs[b*T+t] @ W{1,2}[:, n] + b{1,2}[n]
// blockIdx.y = z selects W1/W2. 64x32 tile per wave.
__global__ __launch_bounds__(256) void gemm_td_kernel(const unsigned short* __restrict__ outs,
                                                      const unsigned short* __restrict__ W1T,
                                                      const unsigned short* __restrict__ W2T,
                                                      const float* __restrict__ b1,
                                                      const float* __restrict__ b2,
                                                      float* __restrict__ dout) {
    const int z = blockIdx.y;
    const unsigned short* WT   = z ? W2T : W1T;
    const float*          bias = z ? b2 : b1;
    float* outp = dout + (size_t)z * Bn * Tn * OUTn;

    const int lane = threadIdx.x & 31;
    const int wave = threadIdx.x >> 5;
    const int half = lane >> 4;
    const int lane16 = lane & 15;

    const int nTN = OUTn / 32;                    // 16 N-tiles
    const int wid = blockIdx.x * 8 + wave;        // 4096 wave tiles
    const int tm = wid / nTN, tn = wid % nTN;
    const int m0 = tm * 64, n0 = tn * 32;

    const unsigned short* arow[4];
    for (int mi = 0; mi < 4; ++mi)
        arow[mi] = outs + (size_t)(m0 + mi * 16 + lane16) * Hn;
    const unsigned short* brow0 = WT + (size_t)(n0 + lane16) * Hn;
    const unsigned short* brow1 = WT + (size_t)(n0 + 16 + lane16) * Hn;

    v8f acc[4][2];
    for (int i = 0; i < 4; ++i)
        for (int j = 0; j < 2; ++j)
            for (int r = 0; r < 8; ++r) acc[i][j][r] = 0.0f;

    for (int kc = 0; kc < Hn; kc += 32) {
        v16bf b0 = load_frag_bf16(brow0, kc, half);
        v16bf b1 = load_frag_bf16(brow1, kc, half);
        for (int mi = 0; mi < 4; ++mi) {
            v16bf a = load_frag_bf16(arow[mi], kc, half);
            acc[mi][0] = wmma_bf16(a, b0, acc[mi][0]);
            acc[mi][1] = wmma_bf16(a, b1, acc[mi][1]);
        }
    }

    for (int mi = 0; mi < 4; ++mi) {
        for (int ni = 0; ni < 2; ++ni) {
            int n = n0 + ni * 16 + lane16;
            float bv = bias[n];
            for (int r = 0; r < 8; ++r) {
                int g = m0 + mi * 16 + r + 8 * half;   // g = b*T + t
                outp[(size_t)g * OUTn + n] = acc[mi][ni][r] + bv;
            }
        }
    }
}

// ---------------------------------------------------------------------------
extern "C" void kernel_launch(void* const* d_in, const int* in_sizes, int n_in,
                              void* d_out, int out_size, void* d_ws, size_t ws_size,
                              hipStream_t stream) {
    (void)in_sizes; (void)n_in; (void)out_size; (void)ws_size;

    const float* x  = (const float*)d_in[0];
    const float* h  = (const float*)d_in[1];
    const float* Wx = (const float*)d_in[2];
    const float* bx = (const float*)d_in[3];
    const float* Wh = (const float*)d_in[4];
    const float* bh = (const float*)d_in[5];
    const float* W1 = (const float*)d_in[6];
    const float* b1 = (const float*)d_in[7];
    const float* W2 = (const float*)d_in[8];
    const float* b2 = (const float*)d_in[9];

    float* dout = (float*)d_out;
    char*  ws   = (char*)d_ws;

    // Workspace layout (bytes).
    const size_t OFF_WXT  = 0;                                    // 1 MB
    const size_t OFF_WHT  = OFF_WXT + (size_t)INn * Hn * 2;       // 2 MB
    const size_t OFF_W1T  = OFF_WHT + (size_t)Hn * Hn * 2;        // 1 MB
    const size_t OFF_W2T  = OFF_W1T + (size_t)Hn * OUTn * 2;      // 1 MB
    const size_t OFF_H0   = OFF_W2T + (size_t)Hn * OUTn * 2;      // 64 KB
    const size_t OFF_XB   = OFF_H0 + (size_t)Bn * Hn * 2;         // 16 MB bf16 x
    const size_t OFF_P    = OFF_XB + (size_t)Bn * Tn * INn * 2;   // 64 MB fp32
    const size_t OFF_OUTS = OFF_P + (size_t)Bn * Tn * Hn * 4;     // 32 MB bf16

    unsigned short* WxT  = (unsigned short*)(ws + OFF_WXT);
    unsigned short* WhT  = (unsigned short*)(ws + OFF_WHT);
    unsigned short* W1T  = (unsigned short*)(ws + OFF_W1T);
    unsigned short* W2T  = (unsigned short*)(ws + OFF_W2T);
    unsigned short* h0   = (unsigned short*)(ws + OFF_H0);
    unsigned short* xb   = (unsigned short*)(ws + OFF_XB);
    float*          P    = (float*)(ws + OFF_P);
    unsigned short* outs = (unsigned short*)(ws + OFF_OUTS);

    // 1) Weight transpose+convert, x convert, h0 convert.
    transpose_cvt_kernel<<<(INn * Hn + 255) / 256, 256, 0, stream>>>(Wx, WxT, INn, Hn);
    transpose_cvt_kernel<<<(Hn * Hn + 255) / 256, 256, 0, stream>>>(Wh, WhT, Hn, Hn);
    transpose_cvt_kernel<<<(Hn * OUTn + 255) / 256, 256, 0, stream>>>(W1, W1T, Hn, OUTn);
    transpose_cvt_kernel<<<(Hn * OUTn + 255) / 256, 256, 0, stream>>>(W2, W2T, Hn, OUTn);
    cvt_kernel<<<((Bn * Tn * INn) + 255) / 256, 256, 0, stream>>>(x, xb, Bn * Tn * INn);
    cvt_kernel<<<(Bn * Hn + 255) / 256, 256, 0, stream>>>(h, h0, Bn * Hn);

    // 2) P = x@Wx + bx + bh   (t-major layout for the recurrent steps)
    //    wave tiles: (16384/64) * (1024/32) = 8192 -> 1024 blocks x 8 waves.
    gemm_xwx_kernel<<<1024, 256, 0, stream>>>(xb, WxT, bx, bh, P);

    // 3) Sequential RNN steps. Step t reads step t-1's bf16 output rows.
    for (int t = 0; t < Tn; ++t) {
        const unsigned short* hp;
        long long stride;
        if (t == 0) { hp = h0; stride = Hn; }
        else        { hp = outs + (size_t)(t - 1) * Hn; stride = (long long)Tn * Hn; }
        float* nh = (t == Tn - 1) ? (dout + 2 * (size_t)Bn * Tn * OUTn) : nullptr;
        rnn_step_kernel<<<16, 128, 0, stream>>>(hp, stride, WhT,
                                                P + (size_t)t * Bn * Hn,
                                                outs, t, nh);
    }

    // 4) td0/td1 GEMMs: (16384/64)*(512/32) = 4096 tiles -> 512 blocks x 8 waves.
    gemm_td_kernel<<<dim3(512, 2), 256, 0, stream>>>(outs, W1T, W2T, b1, b2, dout);
}
